// SynthesizerAttention_41154376631122
// MI455X (gfx1250) — matrix operand order repaired and checked by reading, not compiled, not run on hardware
//
#include <hip/hip_runtime.h>
#include <hip/hip_bf16.h>

// ---------------------------------------------------------------------------
// Synthesizer attention for MI455X (gfx1250, wave32, WMMA bf16, async-to-LDS)
// B=2, T=2048, C=1024, H=16, HD=64
// ---------------------------------------------------------------------------

typedef __attribute__((ext_vector_type(16))) __bf16 v16bf;
typedef __attribute__((ext_vector_type(8)))  float  v8f;
typedef __attribute__((ext_vector_type(4)))  unsigned int u32x4_t;
typedef __attribute__((ext_vector_type(4)))  int i32x4_t;

#define GLOBAL_AS __attribute__((address_space(1)))
#define LDS_AS    __attribute__((address_space(3)))

union FragB {            // 16 bf16 operand fragment = 8 VGPRs of packed pairs
    v16bf v;
    unsigned int u[8];
};

#if defined(__has_builtin)
# if __has_builtin(__builtin_amdgcn_global_load_async_to_lds_b128)
#  define USE_ASYNC_LDS 1
# endif
#endif
#ifndef USE_ASYNC_LDS
# define USE_ASYNC_LDS 0
#endif

// copy 16 bytes global -> LDS (async DMA if available, else through VGPRs)
__device__ __forceinline__ void copy16(const unsigned short* g, unsigned short* l) {
#if USE_ASYNC_LDS
    __builtin_amdgcn_global_load_async_to_lds_b128(
        (GLOBAL_AS i32x4_t*)(g),
        (LDS_AS    i32x4_t*)(l), 0, 0);
#else
    *(u32x4_t*)l = *(const u32x4_t*)g;
#endif
}

__device__ __forceinline__ void wait_async0() {
#if USE_ASYNC_LDS
    asm volatile("s_wait_asynccnt 0x0" ::: "memory");
#endif
}

__device__ __forceinline__ unsigned short f2bf(float f) {
    union { float f; unsigned int u; } v; v.f = f;
    unsigned int r = v.u + 0x7FFFu + ((v.u >> 16) & 1u);   // round-to-nearest-even
    return (unsigned short)(r >> 16);
}

// ---------------------------------------------------------------------------
// elementwise fp32 -> bf16 cast
// ---------------------------------------------------------------------------
__global__ void cast_bf16_kernel(const float* __restrict__ in,
                                 unsigned short* __restrict__ out, int n) {
    int i = blockIdx.x * blockDim.x + threadIdx.x;
    if (i < n) out[i] = f2bf(in[i]);
}

// w2 [64][2048] fp32 -> w2t bf16 [2048][64]:  w2t[key][d] = w2[d][key]
__global__ void w2_transpose_cast_kernel(const float* __restrict__ w2,
                                         unsigned short* __restrict__ w2t) {
    int i = blockIdx.x * blockDim.x + threadIdx.x;
    if (i < 2048 * 64) {
        int key = i >> 6, d = i & 63;
        w2t[i] = f2bf(w2[d * 2048 + key]);
    }
}

// ---------------------------------------------------------------------------
// "AT" GEMM:  Dt[n][m] = sum_k W[n][k] * X[m][k] + bias[n]
//   W: A-operand [N][K] row-major (torch Linear weight, used as-is)
//   X: B-operand [M][K] row-major
// Both LDS tiles are straight row copies -> async-to-LDS, double buffered.
// Block: 256 thr = 8 waves; tile N=64, M=256; wave = 64n x 32m; K-step 32.
// STORE_NM: Out[n*ldo+m]   else Out[m*ldo+n].
// ---------------------------------------------------------------------------
template <bool RELU, bool OUT_BF16, bool STORE_NM>
__global__ __launch_bounds__(256)
void gemm_at_kernel(const unsigned short* __restrict__ W, int lda,
                    const unsigned short* __restrict__ X, int ldb,
                    const float* __restrict__ bias,
                    void* __restrict__ Out, int ldo,
                    int N, int M, int K) {
    // pitch 40 bf16 = 80B (16B multiple -> async segment alignment holds)
    __shared__ __align__(16) unsigned short sW[2][64 * 40];
    __shared__ __align__(16) unsigned short sX[2][256 * 40];
    __shared__ __align__(16) float sBias[64];

    const int tid  = threadIdx.x;
    const int lane = tid & 31;
    const int wave = tid >> 5;
    const int lh   = lane >> 4;
    const int ln   = lane & 15;
    const int n0   = blockIdx.x * 64;
    const int m0   = blockIdx.y * 256;

    const v8f zero = {0.f,0.f,0.f,0.f,0.f,0.f,0.f,0.f};
    v8f acc[4][2];
#pragma unroll
    for (int nt = 0; nt < 4; ++nt) { acc[nt][0] = zero; acc[nt][1] = zero; }

    if (tid < 64) sBias[tid] = bias[n0 + tid];

    const int nsteps = K >> 5;
    // issue tile 0 into buffer 0
    for (int p = tid; p < 1280; p += 256) {
        if (p < 256) {
            int row = p >> 2, seg = p & 3;
            copy16(W + (size_t)(n0 + row) * lda + seg * 8,
                   &sW[0][row * 40 + seg * 8]);
        } else {
            int q = p - 256; int row = q >> 2, seg = q & 3;
            copy16(X + (size_t)(m0 + row) * ldb + seg * 8,
                   &sX[0][row * 40 + seg * 8]);
        }
    }

    for (int ks = 0; ks < nsteps; ++ks) {
        wait_async0();
        __syncthreads();                 // tile ks visible; prev compute done
        if (ks + 1 < nsteps) {           // prefetch tile ks+1 into other buffer
            const int k0 = (ks + 1) << 5;
            const int buf = (ks + 1) & 1;
            for (int p = tid; p < 1280; p += 256) {
                if (p < 256) {
                    int row = p >> 2, seg = p & 3;
                    copy16(W + (size_t)(n0 + row) * lda + k0 + seg * 8,
                           &sW[buf][row * 40 + seg * 8]);
                } else {
                    int q = p - 256; int row = q >> 2, seg = q & 3;
                    copy16(X + (size_t)(m0 + row) * ldb + k0 + seg * 8,
                           &sX[buf][row * 40 + seg * 8]);
                }
            }
        }
        const int cur = ks & 1;
        const unsigned int* sw32 = (const unsigned int*)sW[cur];
        const unsigned int* sx32 = (const unsigned int*)sX[cur];

        // B fragments: this wave's 32 x-rows (pairs over k adjacent)
        FragB xb[2];
#pragma unroll
        for (int bi = 0; bi < 2; ++bi)
#pragma unroll
            for (int i = 0; i < 8; ++i)
                xb[bi].u[i] = sx32[(wave * 32 + bi * 16 + ln) * 20 + (lh << 3) + i];
#pragma unroll
        for (int nt = 0; nt < 4; ++nt) {
            FragB wa;      // A fragment: 16 weight rows
#pragma unroll
            for (int i = 0; i < 8; ++i) {
                int Kl = (lh << 4) + (i << 1);
                wa.u[i] = sw32[(nt * 16 + ln) * 20 + (Kl >> 1)];
            }
            acc[nt][0] = __builtin_amdgcn_wmma_f32_16x16x32_bf16(
                false, wa.v, false, xb[0].v, (short)0, acc[nt][0], false, false);
            acc[nt][1] = __builtin_amdgcn_wmma_f32_16x16x32_bf16(
                false, wa.v, false, xb[1].v, (short)0, acc[nt][1], false, false);
        }
    }

    // epilogue: D value at (n = n0+nt*16+r+8*lh, m = m0+wave*32+bi*16+ln)
#pragma unroll
    for (int nt = 0; nt < 4; ++nt)
#pragma unroll
        for (int bi = 0; bi < 2; ++bi) {
            int m = m0 + wave * 32 + bi * 16 + ln;
#pragma unroll
            for (int r = 0; r < 8; ++r) {
                int nl = nt * 16 + r + (lh << 3);
                int n  = n0 + nl;
                float v = acc[nt][bi][r] + sBias[nl];
                if (RELU) v = fmaxf(v, 0.0f);
                size_t idx = STORE_NM ? (size_t)n * ldo + m : (size_t)m * ldo + n;
                if (OUT_BF16) ((unsigned short*)Out)[idx] = f2bf(v);
                else          ((float*)Out)[idx] = v;
            }
        }
}

// ---------------------------------------------------------------------------
// Fused synthesizer attention, S^T orientation (flash-style online softmax).
// grid = (T/128, B*H); block = 256 (8 waves); wave owns 16 query columns t.
//   S^T[key][t] = w2t[key][:] . a[t][:]          (A=w2t tile, B=a rows, WMMA)
//   softmax per column t (per-lane scalars m,l; one shfl_xor(16))
//   O^T[d][t] += V^T[d][keys] . P[t][keys]       (A=v_t tile, B=P from LDS)
// All tile stagings are straight copies -> async-to-LDS, double buffered.
// ---------------------------------------------------------------------------
__global__ __launch_bounds__(256)
void synth_attn_kernel(const unsigned short* __restrict__ a_bf,
                       const unsigned short* __restrict__ w2t,
                       const float* __restrict__ b2,
                       const unsigned short* __restrict__ v_t,
                       unsigned short* __restrict__ y_bf) {
    constexpr int T = 2048, C = 1024, MT = 4096, HD = 64;

    __shared__ __align__(16) unsigned short sW2t[2][32 * 72]; // [key][d] pitch 72
    __shared__ __align__(16) unsigned short sVt[2][64 * 40];  // [d][key] pitch 40
    __shared__ __align__(16) unsigned short sP[8 * 16 * 34];  // per-wave P [t][key]
    __shared__ __align__(16) float sB2[2][32];

    const int tid  = threadIdx.x;
    const int lane = tid & 31;
    const int wave = tid >> 5;
    const int lh   = lane >> 4;
    const int ln   = lane & 15;
    const int q0   = blockIdx.x * 128;
    const int bh   = blockIdx.y;
    const int b    = bh >> 4;
    const int h    = bh & 15;
    const int t    = q0 + wave * 16 + ln;      // this lane's query column

    // B operand: this wave's 16 rows of `a` (two 16x32-k fragments, kept in regs)
    FragB af[2];
    {
        const unsigned int* ag32 = (const unsigned int*)a_bf;
        size_t rowbase = ((size_t)(b * T + t) * C + h * HD) >> 1;
#pragma unroll
        for (int s = 0; s < 2; ++s)
#pragma unroll
            for (int i = 0; i < 8; ++i)
                af[s].u[i] = ag32[rowbase + s * 16 + (lh << 3) + i];
    }

    const v8f zero = {0.f,0.f,0.f,0.f,0.f,0.f,0.f,0.f};
    float mrun = -1e30f, lrun = 0.0f;
    v8f o[4];
#pragma unroll
    for (int i = 0; i < 4; ++i) o[i] = zero;

    unsigned int* sP32 = (unsigned int*)sP;
    const int tiles = (q0 + 128) >> 5;         // uniform per block

    // issue tile 0
    for (int p = tid; p < 520; p += 256) {
        if (p < 256) {
            int row = p >> 3, seg = p & 7;
            copy16(w2t + (size_t)row * HD + seg * 8,
                   &sW2t[0][row * 72 + seg * 8]);
        } else if (p < 512) {
            int q = p - 256; int row = q >> 2, seg = q & 3;
            copy16(v_t + (size_t)(h * HD + row) * MT + b * T + seg * 8,
                   &sVt[0][row * 40 + seg * 8]);
        } else {
            int j = p - 512;
            copy16((const unsigned short*)b2 + j * 8,
                   (unsigned short*)&sB2[0][0] + j * 8);
        }
    }

    for (int kt = 0; kt < tiles; ++kt) {
        const int k0 = kt << 5;
        wait_async0();
        __syncthreads();
        if (kt + 1 < tiles) {
            const int kn = (kt + 1) << 5;
            const int buf = (kt + 1) & 1;
            for (int p = tid; p < 520; p += 256) {
                if (p < 256) {
                    int row = p >> 3, seg = p & 7;
                    copy16(w2t + (size_t)(kn + row) * HD + seg * 8,
                           &sW2t[buf][row * 72 + seg * 8]);
                } else if (p < 512) {
                    int q = p - 256; int row = q >> 2, seg = q & 3;
                    copy16(v_t + (size_t)(h * HD + row) * MT + b * T + kn + seg * 8,
                           &sVt[buf][row * 40 + seg * 8]);
                } else {
                    int j = p - 512;
                    copy16((const unsigned short*)(b2 + kn) + j * 8,
                           (unsigned short*)&sB2[buf][0] + j * 8);
                }
            }
        }
        const int cur = kt & 1;
        const unsigned int* sw2 = (const unsigned int*)sW2t[cur];
        const unsigned int* svt = (const unsigned int*)sVt[cur];

        // S^T tiles: c0 keys k0+0..15, c1 keys k0+16..31 (rows), cols = t
        v8f c0 = zero, c1 = zero;
#pragma unroll
        for (int s = 0; s < 2; ++s) {
            FragB wa0, wa1;
#pragma unroll
            for (int i = 0; i < 8; ++i) {
                int Kl = s * 32 + (lh << 4) + (i << 1);
                wa0.u[i] = sw2[ln * 36 + (Kl >> 1)];
                wa1.u[i] = sw2[(16 + ln) * 36 + (Kl >> 1)];
            }
            c0 = __builtin_amdgcn_wmma_f32_16x16x32_bf16(
                false, wa0.v, false, af[s].v, (short)0, c0, false, false);
            c1 = __builtin_amdgcn_wmma_f32_16x16x32_bf16(
                false, wa1.v, false, af[s].v, (short)0, c1, false, false);
        }

        // per-column (per-lane) online softmax
        float sv[16];
        float mx = -1e30f;
#pragma unroll
        for (int r = 0; r < 8; ++r) {
            int kl = r + (lh << 3);
            float s0 = c0[r] + sB2[cur][kl];
            float s1 = c1[r] + sB2[cur][16 + kl];
            if (k0 + kl      > t) s0 = -1e10f;
            if (k0 + 16 + kl > t) s1 = -1e10f;
            sv[r] = s0; sv[8 + r] = s1;
            mx = fmaxf(mx, fmaxf(s0, s1));
        }
        mx = fmaxf(mx, __shfl_xor(mx, 16, 32));   // combine the two key-halves
        float nm = fmaxf(mrun, mx);
        float sc = __expf(mrun - nm);
        mrun = nm;
        float rs = 0.0f;
        float pv[16];
#pragma unroll
        for (int j = 0; j < 16; ++j) { pv[j] = __expf(sv[j] - nm); rs += pv[j]; }
        rs += __shfl_xor(rs, 16, 32);
        lrun = lrun * sc + rs;
#pragma unroll
        for (int i = 0; i < 4; ++i) o[i] = o[i] * sc;   // uniform per-lane rescale

        // P -> LDS [t][key] (row-major over key, wave-private region)
        int prow = (wave * 16 + ln) * 34;
#pragma unroll
        for (int r = 0; r < 8; ++r) {
            sP[prow + r + (lh << 3)]      = f2bf(pv[r]);
            sP[prow + 16 + r + (lh << 3)] = f2bf(pv[8 + r]);
        }
        asm volatile("s_wait_dscnt 0x0" ::: "memory");  // same-wave LDS RAW

        // O^T += V^T @ P : A = v_t tile rows d, B = P cols t
        FragB pb;
#pragma unroll
        for (int i = 0; i < 8; ++i)
            pb.u[i] = sP32[(wave * 16 + ln) * 17 + (lh << 3) + i];
#pragma unroll
        for (int dnt = 0; dnt < 4; ++dnt) {
            FragB va;
#pragma unroll
            for (int i = 0; i < 8; ++i) {
                int Kl = (lh << 4) + (i << 1);
                va.u[i] = svt[(dnt * 16 + ln) * 20 + (Kl >> 1)];
            }
            o[dnt] = __builtin_amdgcn_wmma_f32_16x16x32_bf16(
                false, va.v, false, pb.v, (short)0, o[dnt], false, false);
        }
    }

    // epilogue: y[b,t, h*64 + d] = O^T[d][t] / l   (lane owns column t)
    float inv = 1.0f / lrun;
    size_t base = (size_t)(b * T + t) * C + h * HD;
#pragma unroll
    for (int dnt = 0; dnt < 4; ++dnt)
#pragma unroll
        for (int r = 0; r < 8; ++r)
            y_bf[base + dnt * 16 + r + (lh << 3)] = f2bf(o[dnt][r] * inv);
}

// ---------------------------------------------------------------------------
// Launch
// ---------------------------------------------------------------------------
extern "C" void kernel_launch(void* const* d_in, const int* in_sizes, int n_in,
                              void* d_out, int out_size, void* d_ws, size_t ws_size,
                              hipStream_t stream) {
    (void)in_sizes; (void)n_in; (void)out_size; (void)ws_size;

    const float* x       = (const float*)d_in[0];
    const float* w1_w    = (const float*)d_in[1];
    const float* w1_b    = (const float*)d_in[2];
    const float* w2      = (const float*)d_in[3];
    const float* b2      = (const float*)d_in[4];
    const float* value_w = (const float*)d_in[5];
    const float* value_b = (const float*)d_in[6];
    const float* proj_w  = (const float*)d_in[7];
    const float* proj_b  = (const float*)d_in[8];
    float* out = (float*)d_out;

    constexpr int Bc = 2, T = 2048, C = 1024, H = 16;
    constexpr size_t MT = (size_t)Bc * T;          // 4096 rows

    // workspace carve-out (bf16 buffers), ~38.25 MiB total
    unsigned short* ws   = (unsigned short*)d_ws;
    unsigned short* x_bf = ws;                          // MT*C
    unsigned short* w1bf = x_bf + MT * C;               // C*C
    unsigned short* vwbf = w1bf + (size_t)C * C;        // C*C
    unsigned short* pwbf = vwbf + (size_t)C * C;        // C*C
    unsigned short* w2t  = pwbf + (size_t)C * C;        // 2048*64
    unsigned short* a_bf = w2t + (size_t)2048 * 64;     // MT*C   a[t][feat]
    unsigned short* v_t  = a_bf + MT * C;               // C*MT   v^T[feat][row]
    unsigned short* y_bf = v_t + (size_t)C * MT;        // MT*C   y[t][feat]

    const int n_x = (int)(MT * C);
    cast_bf16_kernel<<<(n_x + 255) / 256, 256, 0, stream>>>(x, x_bf, n_x);
    cast_bf16_kernel<<<(C * C + 255) / 256, 256, 0, stream>>>(w1_w, w1bf, C * C);
    cast_bf16_kernel<<<(C * C + 255) / 256, 256, 0, stream>>>(value_w, vwbf, C * C);
    cast_bf16_kernel<<<(C * C + 255) / 256, 256, 0, stream>>>(proj_w, pwbf, C * C);
    w2_transpose_cast_kernel<<<(2048 * 64 + 255) / 256, 256, 0, stream>>>(w2, w2t);

    dim3 ggrid(C / 64, (unsigned)(MT / 256));   // (16, 16)
    // a[t][n] = relu(x @ w1^T + b1)            (store m,n)
    gemm_at_kernel<true,  true,  false><<<ggrid, 256, 0, stream>>>(
        w1bf, C, x_bf, C, w1_b, a_bf, C, C, (int)MT, C);
    // v^T[n][t] = (x @ value^T + bv)^T         (store n,m)
    gemm_at_kernel<false, true,  true ><<<ggrid, 256, 0, stream>>>(
        vwbf, C, x_bf, C, value_b, v_t, (int)MT, C, (int)MT, C);
    // fused scores/softmax/att@v -> y[t][feat] (bf16)
    synth_attn_kernel<<<dim3(T / 128, Bc * H), 256, 0, stream>>>(
        a_bf, w2t, b2, v_t, y_bf);
    // out[t][n] = y @ proj^T + bp              (f32, store m,n)
    gemm_at_kernel<false, false, false><<<ggrid, 256, 0, stream>>>(
        pwbf, C, y_bf, C, proj_b, out, C, C, (int)MT, C);
}
